// MultiHeadSelfAttention_798863917504
// MI455X (gfx1250) — matrix operand order repaired and checked by reading, not compiled, not run on hardware
//
#include <hip/hip_runtime.h>
#include <hip/hip_bf16.h>

typedef __bf16 bf16_t;
typedef __attribute__((ext_vector_type(16))) __bf16 v16bf;
typedef __attribute__((ext_vector_type(2)))  __bf16 v2bf;
typedef __attribute__((ext_vector_type(8)))  float  v8f;
typedef __attribute__((ext_vector_type(4)))  unsigned int v4u;
typedef __attribute__((ext_vector_type(8)))  int    v8i;
typedef __attribute__((ext_vector_type(4)))  int    v4i;

#define B_   64
#define N_   50
#define W_   30
#define D_   300
#define H_   16
#define O_   16
#define WP   32          // padded W
#define DP   320         // padded D
#define BN_  (B_ * N_)   // 3200 sequences
#define GPB  8           // sequences per block (one per wave)
#define KST  (DP / 32)   // 10 k-steps of 32
#define NCHUNK (H_ * KST)
#define CHUNK_ELEMS (32 * DP)   // one Q e-chunk: 32 e-rows x 320 d

// ---------------- WMMA helpers (bf16 16x16x32, f32 accum) ----------------

__device__ __forceinline__ v8f wmma_bf16(v16bf a, v16bf b, v8f c) {
  // (neg_a, A, neg_b, B, c_mod, C, reuse_a, reuse_b)
  return __builtin_amdgcn_wmma_f32_16x16x32_bf16(false, a, false, b, (short)0, c,
                                                 false, false);
}

// A fragment: 16x32 bf16 from row-major [16 rows, ldk cols].
// lane L (0-15): row L, k in {0..7,16..23}; lane L+16: row L, k in {8..15,24..31}
__device__ __forceinline__ v16bf load_frag_a(const bf16_t* base, int ldk) {
  const int l = threadIdx.x & 31;
  const bf16_t* row = base + (l & 15) * ldk + ((l >> 4) << 3);
  v16bf a;
#pragma unroll
  for (int v = 0; v < 8; ++v) {
    const int k = ((v & 3) << 1) + ((v >> 2) << 4);
    v2bf p = *(const v2bf*)(row + k);
    a[2 * v] = p.x;
    a[2 * v + 1] = p.y;
  }
  return a;
}

// B fragment: 32(K)x16(N) bf16 where source is stored transposed row-major
// [16 n-rows, ldk k-cols] (i.e. srcT[n][k] == B[k][n]).
// lanes 0-15: n=lane, K in {0..7,16..23}; lanes 16-31: n=lane-16, K in {8..15,24..31}
__device__ __forceinline__ v16bf load_frag_b(const bf16_t* baseT, int ldk) {
  const int l = threadIdx.x & 31;
  const bf16_t* row = baseT + (l & 15) * ldk + ((l >> 4) << 3);
  v16bf b;
#pragma unroll
  for (int v = 0; v < 8; ++v) {
    v2bf p = *(const v2bf*)(row + ((v & 3) << 1) + ((v >> 2) << 4));
    b[2 * v] = p.x;
    b[2 * v + 1] = p.y;
  }
  return b;
}

// ---------------- TDM: async load one Q e-chunk (32 x 320 bf16) into LDS ----

__device__ __forceinline__ void tdm_load_q_chunk(const bf16_t* gsrc,
                                                 unsigned lds_off) {
  unsigned long long ga = (unsigned long long)(uintptr_t)gsrc;
  v4u g0;
  g0[0] = 1u;                                   // count=1, user descriptor
  g0[1] = lds_off;                              // lds_addr (bytes)
  g0[2] = (unsigned)(ga & 0xFFFFFFFFu);         // global_addr[31:0]
  g0[3] = (unsigned)((ga >> 32) & 0x01FFFFFFu)  // global_addr[56:32]
          | (2u << 30);                         // type=2 ("image")
  v8i g1;
  g1[0] = (1 << 16);                 // workgroup_mask=0, data_size=1 (2 bytes)
  g1[1] = (DP & 0xFFFF) << 16;       // tensor_dim0 = 320 (bits 79:48, low half)
  g1[2] = (DP >> 16) | (32 << 16);   // tensor_dim0 hi, tensor_dim1 = 32
  g1[3] = (DP << 16);                // tensor_dim1 hi = 0, tile_dim0 = 320
  g1[4] = 32;                        // tile_dim1 = 32, tile_dim2 = 0
  g1[5] = DP;                        // tensor_dim0_stride = 320 (low 32)
  g1[6] = 0;                         // stride hi, tensor_dim1_stride lo
  g1[7] = 0;
  v4i g2 = {0, 0, 0, 0};
  v4i g3 = {0, 0, 0, 0};
#if __has_include(<hip/amd_detail/amd_gfx1250_TDM.h>)
  v8i g4 = {0, 0, 0, 0, 0, 0, 0, 0};
  __builtin_amdgcn_tensor_load_to_lds(g0, g1, g2, g3, g4, 0);
#else
  __builtin_amdgcn_tensor_load_to_lds(g0, g1, g2, g3, 0);
#endif
}

// ---------------- prep: Q[h][d][e] f32 -> Qt[h][e][d] bf16 (zero padded) ----

__global__ __launch_bounds__(256) void qt_prep_kernel(
    const float* __restrict__ Q, bf16_t* __restrict__ Qt) {
  int idx = blockIdx.x * blockDim.x + threadIdx.x;
  if (idx >= H_ * DP * DP) return;
  int d = idx % DP;
  int e = (idx / DP) % DP;
  int h = idx / (DP * DP);
  float v = 0.0f;
  if (d < D_ && e < D_) v = Q[(h * D_ + d) * D_ + e];
  Qt[idx] = (bf16_t)v;
}

// ---------------- main fused attention kernel ----------------

__global__ __launch_bounds__(256, 1) void mhsa_kernel(
    const float* __restrict__ X,      // [BN, W, D]
    const float* __restrict__ Vg,     // [H, O, D]
    const bf16_t* __restrict__ Qt,    // [H, DP(e), DP(d)] bf16 (prepped)
    float* __restrict__ out) {        // [BN, W, H*O]
  __shared__ bf16_t s_x[GPB][WP][DP];     // per-wave x, bf16, padded  (160 KB)
  __shared__ bf16_t s_q[2][32][DP];       // double-buffered Q chunk   (40 KB)
  __shared__ bf16_t s_v[O_][DP];          // V_h transposed-friendly   (10 KB)
  __shared__ float  s_alpha[GPB][WP][WP]; // per-wave logits           (32 KB)
  __shared__ bf16_t s_xq[GPB][WP][WP];    // per-wave xq / attn (reuse)(16 KB)
  __shared__ bf16_t s_xvT[GPB][O_][WP];   // per-wave (x V^T)^T        (8 KB)

  const int wave = threadIdx.x >> 5;
  const int lane = threadIdx.x & 31;
  const int bn = blockIdx.x * GPB + wave;
  const int lo = lane & 15;          // N index inside a 16-wide tile
  const int hi8 = (lane >> 4) << 3;  // +8 row offset for upper half-wave

  // ---- stage this wave's x into LDS as zero-padded bf16 [32][320] ----
  {
    const float* xg = X + (size_t)bn * W_ * D_;
    for (int idx = lane; idx < WP * DP; idx += 32) {
      int w = idx / DP, d = idx % DP;
      float v = (w < W_ && d < D_) ? xg[w * D_ + d] : 0.0f;
      s_x[wave][w][d] = (bf16_t)v;
    }
  }

  // ---- prefetch first Q chunk via TDM (wave 0 only) ----
  if (threadIdx.x < 32)
    tdm_load_q_chunk(Qt, (unsigned)(uintptr_t)&s_q[0][0][0]);

  const bf16_t* xw = &s_x[wave][0][0];

  for (int h = 0; h < H_; ++h) {
    __syncthreads();  // prior head done reading s_v
    // ---- stage V_h cooperatively: s_v[o][d] = V[h][o][d] (bf16, padded) ----
    for (int idx = threadIdx.x; idx < O_ * DP; idx += 256) {
      int o = idx / DP, d = idx % DP;
      float v = (d < D_) ? Vg[((size_t)h * O_ + o) * D_ + d] : 0.0f;
      s_v[o][d] = (bf16_t)v;
    }
    __syncthreads();

    v8f acc_al[2][2];
#pragma unroll
    for (int i = 0; i < 2; ++i)
#pragma unroll
      for (int j = 0; j < 2; ++j)
#pragma unroll
        for (int v = 0; v < 8; ++v) acc_al[i][j][v] = 0.0f;

    // ---- fused: xq_c = x @ Q[:, e:e+32]; alpha += xq_c @ x^T[e:e+32] ----
    for (int ec = 0; ec < KST; ++ec) {
      const int c = h * KST + ec;
      if (threadIdx.x < 32) __builtin_amdgcn_s_wait_tensorcnt(0);
      __syncthreads();  // chunk c visible to all waves
      if (threadIdx.x < 32 && (c + 1) < NCHUNK)
        tdm_load_q_chunk(Qt + (size_t)(c + 1) * CHUNK_ELEMS,
                         (unsigned)(uintptr_t)&s_q[(c + 1) & 1][0][0]);

      const bf16_t* qc = &s_q[c & 1][0][0];
      v8f aq[2][2];
#pragma unroll
      for (int i = 0; i < 2; ++i)
#pragma unroll
        for (int j = 0; j < 2; ++j)
#pragma unroll
          for (int v = 0; v < 8; ++v) aq[i][j][v] = 0.0f;

#pragma unroll
      for (int kk = 0; kk < KST; ++kk) {
        v16bf a0 = load_frag_a(xw + kk * 32, DP);             // w rows 0-15
        v16bf a1 = load_frag_a(xw + 16 * DP + kk * 32, DP);   // w rows 16-31
        v16bf b0 = load_frag_b(qc + kk * 32, DP);             // e cols 0-15
        v16bf b1 = load_frag_b(qc + 16 * DP + kk * 32, DP);   // e cols 16-31
        aq[0][0] = wmma_bf16(a0, b0, aq[0][0]);
        aq[0][1] = wmma_bf16(a0, b1, aq[0][1]);
        aq[1][0] = wmma_bf16(a1, b0, aq[1][0]);
        aq[1][1] = wmma_bf16(a1, b1, aq[1][1]);
      }
      // spill xq_c to LDS as bf16 row-major [32][32]
#pragma unroll
      for (int i = 0; i < 2; ++i)
#pragma unroll
        for (int j = 0; j < 2; ++j)
#pragma unroll
          for (int v = 0; v < 8; ++v)
            s_xq[wave][i * 16 + v + hi8][j * 16 + lo] = (bf16_t)aq[i][j][v];
      asm volatile("s_wait_dscnt 0" ::: "memory");

      // alpha += xq_c (32x32) @ x^T chunk (32x32)
      v16bf a0 = load_frag_a(&s_xq[wave][0][0], WP);
      v16bf a1 = load_frag_a(&s_xq[wave][16][0], WP);
      v16bf b0 = load_frag_b(xw + ec * 32, DP);           // B[k=e][n=v]=x[v][e]
      v16bf b1 = load_frag_b(xw + 16 * DP + ec * 32, DP);
      acc_al[0][0] = wmma_bf16(a0, b0, acc_al[0][0]);
      acc_al[0][1] = wmma_bf16(a0, b1, acc_al[0][1]);
      acc_al[1][0] = wmma_bf16(a1, b0, acc_al[1][0]);
      acc_al[1][1] = wmma_bf16(a1, b1, acc_al[1][1]);
      __syncthreads();  // all waves done with s_q[c&1] before c+2 overwrites
    }

    // ---- softmax over rows of alpha (mask padded cols), attn -> s_xq ----
#pragma unroll
    for (int i = 0; i < 2; ++i)
#pragma unroll
      for (int j = 0; j < 2; ++j)
#pragma unroll
        for (int v = 0; v < 8; ++v)
          s_alpha[wave][i * 16 + v + hi8][j * 16 + lo] = acc_al[i][j][v];
    asm volatile("s_wait_dscnt 0" ::: "memory");
    {
      const int r = lane;  // one row per lane (rows >= W_ are harmless junk)
      float vals[WP];
      float mx = -3.0e38f;
#pragma unroll
      for (int c2 = 0; c2 < WP; ++c2) {
        float t = (c2 < W_) ? s_alpha[wave][r][c2] : -3.0e38f;
        vals[c2] = t;
        mx = fmaxf(mx, t);
      }
      float sum = 0.0f;
#pragma unroll
      for (int c2 = 0; c2 < WP; ++c2) {
        float e = __expf(vals[c2] - mx);
        vals[c2] = e;
        sum += e;
      }
      float inv = 1.0f / sum;
#pragma unroll
      for (int c2 = 0; c2 < WP; ++c2)
        s_xq[wave][r][c2] = (bf16_t)(vals[c2] * inv);  // reuse s_xq as attn
    }
    asm volatile("s_wait_dscnt 0" ::: "memory");

    // ---- xv = x @ V_h^T : [32,16], K = 320 ----
    v8f axv[2];
#pragma unroll
    for (int i = 0; i < 2; ++i)
#pragma unroll
      for (int v = 0; v < 8; ++v) axv[i][v] = 0.0f;
#pragma unroll
    for (int kk = 0; kk < KST; ++kk) {
      v16bf a0 = load_frag_a(xw + kk * 32, DP);
      v16bf a1 = load_frag_a(xw + 16 * DP + kk * 32, DP);
      v16bf b0 = load_frag_b(&s_v[0][0] + kk * 32, DP);  // B[k=d][n=o]=V[o][d]
      axv[0] = wmma_bf16(a0, b0, axv[0]);
      axv[1] = wmma_bf16(a1, b0, axv[1]);
    }
    // store transposed for the final B operand: s_xvT[o][w]
#pragma unroll
    for (int i = 0; i < 2; ++i)
#pragma unroll
      for (int v = 0; v < 8; ++v)
        s_xvT[wave][lo][i * 16 + v + hi8] = (bf16_t)axv[i][v];
    asm volatile("s_wait_dscnt 0" ::: "memory");

    // ---- out = attn (32x32) @ xv (32x16); padded attn cols are exactly 0 ----
    {
      v8f o0, o1;
#pragma unroll
      for (int v = 0; v < 8; ++v) { o0[v] = 0.0f; o1[v] = 0.0f; }
      v16bf a0 = load_frag_a(&s_xq[wave][0][0], WP);
      v16bf a1 = load_frag_a(&s_xq[wave][16][0], WP);
      v16bf b0 = load_frag_b(&s_xvT[wave][0][0], WP);
      o0 = wmma_bf16(a0, b0, o0);
      o1 = wmma_bf16(a1, b0, o1);

      float* op = out + (size_t)bn * W_ * (H_ * O_) + h * O_ + lo;
#pragma unroll
      for (int v = 0; v < 8; ++v) {
        int w0 = v + hi8;
        if (w0 < W_) op[(size_t)w0 * (H_ * O_)] = o0[v];
        int w1 = 16 + v + hi8;
        if (w1 < W_) op[(size_t)w1 * (H_ * O_)] = o1[v];
      }
    }
  }
}

// ---------------- host-side launch ----------------

extern "C" void kernel_launch(void* const* d_in, const int* in_sizes, int n_in,
                              void* d_out, int out_size, void* d_ws,
                              size_t ws_size, hipStream_t stream) {
  const float* X = (const float*)d_in[0];   // [B,N,W,D] f32
  const float* Q = (const float*)d_in[1];   // [H,D,D]   f32
  const float* V = (const float*)d_in[2];   // [H,O,D]   f32
  float* out = (float*)d_out;               // [B,N,W,H*O] f32
  bf16_t* Qt = (bf16_t*)d_ws;               // [H,DP,DP] bf16 (3.28 MB)

  {
    int total = H_ * DP * DP;
    qt_prep_kernel<<<(total + 255) / 256, 256, 0, stream>>>(Q, Qt);
  }
  mhsa_kernel<<<BN_ / GPB, 256, 0, stream>>>(X, V, Qt, out);
}
